// MultiHeadSelfAttention_52682068853213
// MI455X (gfx1250) — matrix operand order repaired and checked by reading, not compiled, not run on hardware
//
#include <hip/hip_runtime.h>

typedef __attribute__((ext_vector_type(16))) _Float16 v16h;
typedef __attribute__((ext_vector_type(8)))  _Float16 v8h;
typedef __attribute__((ext_vector_type(4)))  _Float16 v4h;
typedef __attribute__((ext_vector_type(8)))  float    v8f;
typedef __attribute__((ext_vector_type(4)))  float    v4f;

// Problem constants
constexpr int NB  = 16;    // batch
constexpr int NHH = 8;     // heads
constexpr int SEQ = 1024;  // H*W
constexpr int CH  = 512;   // channels
constexpr int DH  = 64;    // dim per head

// ---------------- workspace layout (bytes) ----------------
constexpr size_t SZ_XH = (size_t)NB * NHH * SEQ * DH * 2;   // 16 MB  f16 [B,NH,S,D]
constexpr size_t SZ_W  = (size_t)NHH * DH * DH * 2;         // 64 KB  f16 [NH,D,D]
constexpr size_t SZ_WO = (size_t)CH * CH * 2;               // 512 KB f16 [C,C]
constexpr size_t SZ_P  = (size_t)NB * NHH * SEQ * DH * 2;   // 16 MB  f16 per Q/K/Vt
constexpr size_t SZ_AO = (size_t)NB * SEQ * CH * 2;         // 16 MB  f16 [B,S,C]

constexpr size_t OFF_XH = 0;
constexpr size_t OFF_WQ = OFF_XH + SZ_XH;
constexpr size_t OFF_WK = OFF_WQ + SZ_W;
constexpr size_t OFF_WV = OFF_WK + SZ_W;
constexpr size_t OFF_WO = OFF_WV + SZ_W;
constexpr size_t OFF_Q  = OFF_WO + SZ_WO;
constexpr size_t OFF_K  = OFF_Q  + SZ_P;
constexpr size_t OFF_VT = OFF_K  + SZ_P;
constexpr size_t OFF_AO = OFF_VT + SZ_P;

// dynamic LDS for k_attn: 16x1024 f32 scores + 16x1024 f16 probabilities
constexpr size_t LDS_SC = (size_t)16 * SEQ * sizeof(float);    // 64 KB
constexpr size_t LDS_PR = (size_t)16 * SEQ * sizeof(_Float16); // 32 KB
constexpr size_t LDS_ATTN = LDS_SC + LDS_PR;                   // 96 KB

// ---------------- WMMA helpers ----------------
__device__ __forceinline__ v8f wmma_f16(v16h a, v16h b, v8f c) {
  // D(f32 16x16) = A(f16 16x32) * B(f16 32x16) + C
  return __builtin_amdgcn_wmma_f32_16x16x32_f16(
      /*neg_a=*/false, a, /*neg_b=*/false, b,
      /*c_mod=*/(short)0, c, /*reuse_a=*/false, /*reuse_b=*/false);
}

// A operand: 16x32 f16 tile from row-major storage (ld in elements).
// Lane m%16 = row; lanes 0-15 carry K={0..7,16..23}, lanes 16-31 K={8..15,24..31}.
__device__ __forceinline__ v16h load_a(const _Float16* __restrict__ p, int ld, int lane) {
  int m  = lane & 15;
  int hi = (lane >> 4) & 1;
  const _Float16* base = p + (size_t)m * ld + hi * 8;
  v8h lo = *(const v8h*)(base);        // K = hi*8 .. hi*8+7
  v8h hh = *(const v8h*)(base + 16);   // K = 16+hi*8 ..
  v16h r;
#pragma unroll
  for (int i = 0; i < 8; ++i) { r[i] = lo[i]; r[i + 8] = hh[i]; }
  return r;
}

// Same striping, but sourced from (LDS) f16 probability strip.
__device__ __forceinline__ v16h load_a_lds16(const _Float16* p, int ld, int lane, int k0) {
  int m  = lane & 15;
  int hi = (lane >> 4) & 1;
  const _Float16* base = p + (size_t)m * ld + k0 + hi * 8;
  v8h lo = *(const v8h*)(base);
  v8h hh = *(const v8h*)(base + 16);
  v16h r;
#pragma unroll
  for (int i = 0; i < 8; ++i) { r[i] = lo[i]; r[i + 8] = hh[i]; }
  return r;
}

// B operand: 32x16 f16 tile, storage is column-major per output column:
// p[n*ld + k] = B[k][n]. Lanes 0-15: K={0..7,16..23}, lanes 16-31: K={8..15,24..31}, N=lane%16.
__device__ __forceinline__ v16h load_b(const _Float16* __restrict__ p, int ld, int lane) {
  int nn   = lane & 15;
  int koff = (lane & 16) >> 1;
  const _Float16* base = p + (size_t)nn * ld + koff;
  v8h lo = *(const v8h*)(base);
  v8h hh = *(const v8h*)(base + 16);
  v16h r;
#pragma unroll
  for (int i = 0; i < 8; ++i) { r[i] = lo[i]; r[i + 8] = hh[i]; }
  return r;
}

// ---------------- kernel 1: x = features + pos_enc -> f16 head-major ----------------
__global__ void __launch_bounds__(256) k_prep_x(const float* __restrict__ f,
                                                const float* __restrict__ pe,
                                                _Float16* __restrict__ xh) {
  int i = blockIdx.x * blockDim.x + threadIdx.x;  // over B*S*C
  if (i >= NB * SEQ * CH) return;
  int c = i % CH;
  int s = (i / CH) % SEQ;
  int b = i / (CH * SEQ);
  float v = f[i] + pe[s * CH + c];
  int h = c / DH, d = c % DH;
  xh[(((size_t)b * NHH + h) * SEQ + s) * DH + d] = (_Float16)v;
}

// ---------------- kernel 2: weight downconvert ----------------
__global__ void __launch_bounds__(256) k_prep_w(const float* __restrict__ Wq,
                                                const float* __restrict__ Wk,
                                                const float* __restrict__ Wv,
                                                const float* __restrict__ Wo,
                                                _Float16* __restrict__ q16,
                                                _Float16* __restrict__ k16,
                                                _Float16* __restrict__ v16,
                                                _Float16* __restrict__ o16) {
  int i = blockIdx.x * blockDim.x + threadIdx.x;
  constexpr int NW = NHH * DH * DH;   // 32768
  if (i < NW) {
    q16[i] = (_Float16)Wq[i];
    k16[i] = (_Float16)Wk[i];
    v16[i] = (_Float16)Wv[i];
  }
  if (i < CH * CH) o16[i] = (_Float16)Wo[i];
}

// ---------------- kernel 3: per-head QKV projection ----------------
// grid = B*NH*(S/64), block = 128 (4 waves, 16 rows each)
__global__ void __launch_bounds__(128) k_qkv(
    const _Float16* __restrict__ xh,
    const _Float16* __restrict__ wq, const _Float16* __restrict__ wk,
    const _Float16* __restrict__ wv,
    const float* __restrict__ bq, const float* __restrict__ bk,
    const float* __restrict__ bv,
    _Float16* __restrict__ Q, _Float16* __restrict__ K,
    _Float16* __restrict__ Vt) {
  int lane = threadIdx.x & 31, wave = threadIdx.x >> 5;
  int bid  = blockIdx.x;
  int sblk = bid & 15;
  int h    = (bid >> 4) & 7;
  int b    = bid >> 7;
  int srow = sblk * 64 + wave * 16;
  size_t headoff = (size_t)(b * NHH + h);

  const _Float16* xbase = xh + (headoff * SEQ + srow) * DH;
  v16h a0 = load_a(xbase, DH, lane);       // K = 0..31  (head dim)
  v16h a1 = load_a(xbase + 32, DH, lane);  // K = 32..63

  int n  = lane & 15;
  int mo = (lane >> 4) * 8;

#pragma unroll
  for (int p = 0; p < 3; ++p) {
    const _Float16* W    = (p == 0) ? wq : (p == 1) ? wk : wv;
    const float*    bias = (p == 0) ? bq : (p == 1) ? bk : bv;
#pragma unroll
    for (int ct = 0; ct < 4; ++ct) {
      int colbase = ct * 16;
      // y = x @ W^T : B[d][e] = W[e][d]; W row-major [e][d] is exactly
      // column-major-per-output-column storage -> load_b with ld = DH.
      const _Float16* wb = W + (size_t)h * DH * DH + (size_t)colbase * DH;
      v8f c = {};
      c = wmma_f16(a0, load_b(wb, DH, lane), c);
      c = wmma_f16(a1, load_b(wb + 32, DH, lane), c);
      float bb = bias[h * DH + colbase + n];
      if (p == 0) {
        _Float16* out = Q + (headoff * SEQ + srow) * DH;
#pragma unroll
        for (int r = 0; r < 8; ++r)
          out[(size_t)(r + mo) * DH + colbase + n] = (_Float16)(c[r] + bb);
      } else if (p == 1) {
        _Float16* out = K + (headoff * SEQ + srow) * DH;
#pragma unroll
        for (int r = 0; r < 8; ++r)
          out[(size_t)(r + mo) * DH + colbase + n] = (_Float16)(c[r] + bb);
      } else {
        // store V transposed: Vt[b,h,d,s]
        _Float16* out = Vt + (headoff * DH + colbase + n) * SEQ + srow;
#pragma unroll
        for (int r = 0; r < 8; ++r) out[r + mo] = (_Float16)(c[r] + bb);
      }
    }
  }
}

// ---------------- kernel 4: attention (scores -> softmax -> attn@V) ----------------
// grid = B*NH*(S/16), block = 128; dynamic LDS = 96 KB (f32 scores + f16 probs)
__global__ void __launch_bounds__(128) k_attn(const _Float16* __restrict__ Q,
                                              const _Float16* __restrict__ Kf,
                                              const _Float16* __restrict__ Vt,
                                              _Float16* __restrict__ AO) {
  extern __shared__ char smem[];
  float*    sc = (float*)smem;                    // [16][SEQ] f32 scores
  _Float16* pr = (_Float16*)(smem + LDS_SC);      // [16][SEQ] f16 probabilities

  int tid = threadIdx.x, lane = tid & 31, wave = tid >> 5;
  int bid = blockIdx.x;
  int qt  = bid & 63;
  int h   = (bid >> 6) & 7;
  int b   = bid >> 9;
  int q0  = qt * 16;
  size_t headoff = (size_t)(b * NHH + h);

  const _Float16* qbase = Q + (headoff * SEQ + q0) * DH;
  v16h a0 = load_a(qbase, DH, lane);
  v16h a1 = load_a(qbase + 32, DH, lane);

  int n  = lane & 15;
  int mo = (lane >> 4) * 8;
  const float scale = 0.125f;   // 1/sqrt(64)

  // scores: wave w covers key columns [w*256, w*256+256)
#pragma unroll
  for (int t = 0; t < 16; ++t) {
    int col0 = wave * 256 + t * 16;
    // B[e][k] = K[k][e]; K row-major [s][d] -> contiguous per-lane along e.
    const _Float16* kb = Kf + (headoff * SEQ + col0) * DH;
    if (t < 15) __builtin_prefetch(kb + 16 * DH, 0, 1);  // next tile -> global_prefetch
    v8f c = {};
    c = wmma_f16(a0, load_b(kb, DH, lane), c);
    c = wmma_f16(a1, load_b(kb + 32, DH, lane), c);
#pragma unroll
    for (int r = 0; r < 8; ++r) sc[(size_t)(r + mo) * SEQ + col0 + n] = c[r] * scale;
  }
  __syncthreads();

  // cooperative row softmax (16 rows x 1024 cols, 8 threads per row), vectorized
  __shared__ float red[16][8];
  __shared__ float rowmax[16];
  __shared__ float rowinv[16];

  int rr = tid >> 3, seg = tid & 7;
  const v4f* rowp4 = (const v4f*)(sc + (size_t)rr * SEQ + seg * 128);
  float m = -3.4e38f;
#pragma unroll 4
  for (int i = 0; i < 32; ++i) {
    v4f x = rowp4[i];
    m = fmaxf(m, fmaxf(fmaxf(x[0], x[1]), fmaxf(x[2], x[3])));
  }
  red[rr][seg] = m;
  __syncthreads();
  if (tid < 16) {
    float mm = -3.4e38f;
    for (int j = 0; j < 8; ++j) mm = fmaxf(mm, red[tid][j]);
    rowmax[tid] = mm;
  }
  __syncthreads();
  float rm = rowmax[rr];
  float ssum = 0.f;
  v4h* prp4 = (v4h*)(pr + (size_t)rr * SEQ + seg * 128);
#pragma unroll 4
  for (int i = 0; i < 32; ++i) {
    v4f x = rowp4[i];
    float e0 = __expf(x[0] - rm);
    float e1 = __expf(x[1] - rm);
    float e2 = __expf(x[2] - rm);
    float e3 = __expf(x[3] - rm);
    v4h ph = {(_Float16)e0, (_Float16)e1, (_Float16)e2, (_Float16)e3};
    prp4[i] = ph;
    ssum += (e0 + e1) + (e2 + e3);
  }
  red[rr][seg] = ssum;
  __syncthreads();
  if (tid < 16) {
    float s2 = 0.f;
    for (int j = 0; j < 8; ++j) s2 += red[tid][j];
    rowinv[tid] = 1.0f / s2;
  }
  __syncthreads();

  // attn @ V: wave w produces output columns [w*16, w*16+16)
  int colbase = wave * 16;
  const _Float16* vb = Vt + (headoff * DH + colbase) * SEQ;
  v8f acc = {};
  for (int kk = 0; kk < SEQ; kk += 32) {
    v16h a  = load_a_lds16(pr, SEQ, lane, kk);   // wide LDS b128 loads, no cvt
    v16h bm = load_b(vb + kk, SEQ, lane);        // Vt[col][k] contiguous along k
    acc = wmma_f16(a, bm, acc);
  }
  _Float16* ob = AO + ((size_t)b * SEQ + q0) * CH + h * DH + colbase;
#pragma unroll
  for (int r = 0; r < 8; ++r)
    ob[(size_t)(r + mo) * CH + n] = (_Float16)(acc[r] * rowinv[r + mo]);
}

// ---------------- kernel 5: output projection  Y = AO @ Wo^T + bo ----------------
// grid = (B*S/16) * (C/256), block = 128; each wave: 16x64 strip, 4 accumulators
__global__ void __launch_bounds__(128) k_oproj(const _Float16* __restrict__ AO,
                                               const _Float16* __restrict__ Wo16,
                                               const float* __restrict__ bo,
                                               float* __restrict__ out) {
  int lane = threadIdx.x & 31, wave = threadIdx.x >> 5;
  int bid = blockIdx.x;
  int mt  = bid & 1023;        // M tile (16 rows)
  int ct  = bid >> 10;         // 0..1 : 256-col slab
  int m0  = mt * 16;
  int colw = ct * 256 + wave * 64;   // this wave's 64-col strip

  const _Float16* abase = AO + (size_t)m0 * CH;
  v8f acc[4] = {{}, {}, {}, {}};
  for (int kk = 0; kk < CH; kk += 32) {
    if (kk + 64 < CH) __builtin_prefetch(abase + kk + 64, 0, 1);
    v16h a = load_a(abase + kk, CH, lane);   // 1 A load feeds 4 WMMAs
#pragma unroll
    for (int j = 0; j < 4; ++j) {
      const _Float16* wb = Wo16 + (size_t)(colw + j * 16) * CH + kk;
      acc[j] = wmma_f16(a, load_b(wb, CH, lane), acc[j]);
    }
  }
  int n  = lane & 15;
  int mo = (lane >> 4) * 8;
#pragma unroll
  for (int j = 0; j < 4; ++j) {
    float bias = bo[colw + j * 16 + n];
#pragma unroll
    for (int r = 0; r < 8; ++r)
      out[(size_t)(m0 + r + mo) * CH + colw + j * 16 + n] = acc[j][r] + bias;
  }
}

// ---------------- host launcher ----------------
extern "C" void kernel_launch(void* const* d_in, const int* in_sizes, int n_in,
                              void* d_out, int out_size, void* d_ws, size_t ws_size,
                              hipStream_t stream) {
  const float* features = (const float*)d_in[0];
  const float* pos_enc  = (const float*)d_in[1];
  const float* Wq = (const float*)d_in[2];
  const float* bq = (const float*)d_in[3];
  const float* Wk = (const float*)d_in[4];
  const float* bk = (const float*)d_in[5];
  const float* Wv = (const float*)d_in[6];
  const float* bv = (const float*)d_in[7];
  const float* Wo = (const float*)d_in[8];
  const float* bo = (const float*)d_in[9];

  char* ws = (char*)d_ws;
  _Float16* xh   = (_Float16*)(ws + OFF_XH);
  _Float16* wq16 = (_Float16*)(ws + OFF_WQ);
  _Float16* wk16 = (_Float16*)(ws + OFF_WK);
  _Float16* wv16 = (_Float16*)(ws + OFF_WV);
  _Float16* wo16 = (_Float16*)(ws + OFF_WO);
  _Float16* Qf   = (_Float16*)(ws + OFF_Q);
  _Float16* Kf   = (_Float16*)(ws + OFF_K);
  _Float16* Vt   = (_Float16*)(ws + OFF_VT);
  _Float16* AO   = (_Float16*)(ws + OFF_AO);

  k_prep_x<<<(NB * SEQ * CH + 255) / 256, 256, 0, stream>>>(features, pos_enc, xh);
  k_prep_w<<<(CH * CH + 255) / 256, 256, 0, stream>>>(Wq, Wk, Wv, Wo,
                                                      wq16, wk16, wv16, wo16);
  k_qkv<<<NB * NHH * (SEQ / 64), 128, 0, stream>>>(xh, wq16, wk16, wv16,
                                                   bq, bk, bv, Qf, Kf, Vt);
  k_attn<<<NB * NHH * (SEQ / 16), 128, (int)LDS_ATTN, stream>>>(Qf, Kf, Vt, AO);
  k_oproj<<<(NB * SEQ / 16) * (CH / 256), 128, 0, stream>>>(AO, wo16, bo, (float*)d_out);
}
